// PointNetSetAbstractionSingleScaleGrouping_386547057063
// MI455X (gfx1250) — matrix operand order repaired and checked by reading, not compiled, not run on hardware
//
#include <hip/hip_runtime.h>
#include <hip/hip_bf16.h>

typedef __attribute__((ext_vector_type(16))) _Float16 v16h;
typedef __attribute__((ext_vector_type(8)))  _Float16 v8h;
typedef __attribute__((ext_vector_type(8)))  float    v8f;

#define BATCH   16
#define N_PTS   4096
#define S_NP    1024            // NPOINT
#define K_S     32              // NSAMPLE
#define R2      0.04f           // RADIUS^2
#define NTOT    ((size_t)BATCH * S_NP * K_S)   // 524288 columns
#define NT_TILES ((int)(NTOT / 16))            // 32768 N-tiles
#define BN_EPS  1e-5f
#define RED_BLOCKS 256

// ---------------------------------------------------------------------------
// 1) Farthest point sampling: one block per batch, dist[] in LDS.
//    Records new_xyz coords into d_out[0 : B*3*S] (layout [B,3,S]).
// ---------------------------------------------------------------------------
__global__ __launch_bounds__(256) void fps_kernel(const float* __restrict__ xyz,
                                                  float* __restrict__ out) {
  __shared__ float dist[N_PTS];
  __shared__ float rv[256];
  __shared__ int   ri[256];
  __shared__ int   s_far;
  const int b = blockIdx.x;
  const int tid = threadIdx.x;
  const float* px = xyz + (size_t)b * 3 * N_PTS;
  const float* py = px + N_PTS;
  const float* pz = py + N_PTS;
  for (int i = tid; i < N_PTS; i += 256) dist[i] = 1e10f;
  if (tid == 0) s_far = 0;
  __syncthreads();
  for (int s = 0; s < S_NP; ++s) {
    const int far = s_far;
    const float cx = px[far], cy = py[far], cz = pz[far];
    if (tid == 0) {   // record BEFORE updating (matches scan semantics)
      out[((size_t)b * 3 + 0) * S_NP + s] = cx;
      out[((size_t)b * 3 + 1) * S_NP + s] = cy;
      out[((size_t)b * 3 + 2) * S_NP + s] = cz;
    }
    float lv = -1.0f; int li = 0;
    for (int i = tid; i < N_PTS; i += 256) {
      const float dx = px[i] - cx, dy = py[i] - cy, dz = pz[i] - cz;
      const float d  = dx * dx + dy * dy + dz * dz;
      const float nd = fminf(dist[i], d);
      dist[i] = nd;
      if (nd > lv) { lv = nd; li = i; }   // strict > keeps first max (tie rule)
    }
    rv[tid] = lv; ri[tid] = li;
    __syncthreads();
    for (int o = 128; o > 0; o >>= 1) {
      if (tid < o) {
        const float ov = rv[tid + o]; const int oi = ri[tid + o];
        if (ov > rv[tid] || (ov == rv[tid] && oi < ri[tid])) { rv[tid] = ov; ri[tid] = oi; }
      }
      __syncthreads();
    }
    if (tid == 0) s_far = ri[0];
    __syncthreads();
  }
}

// ---------------------------------------------------------------------------
// 2) Ball query + grouping. One wave32 per center; ballot + prefix-popcount
//    gives ordered slot assignment. x0 layout: [j][16] f16, j = ((b*S+s)*32+k).
//    ch 0..2 rel xyz, 3..5 features, 6..15 zero pad (so WMMA K=32 works with
//    hi-half zero fragment). One 32B v16h store per slot.
// ---------------------------------------------------------------------------
__global__ __launch_bounds__(256) void ball_group_kernel(const float* __restrict__ xyz,
                                                         const float* __restrict__ feat,
                                                         const float* __restrict__ newxyz,
                                                         _Float16* __restrict__ x0) {
  const int gw   = blockIdx.x * 8 + (threadIdx.x >> 5);
  const int lane = threadIdx.x & 31;
  if (gw >= BATCH * S_NP) return;
  const int b = gw / S_NP, s = gw % S_NP;
  const float* px = xyz + (size_t)b * 3 * N_PTS;
  const float* py = px + N_PTS;
  const float* pz = py + N_PTS;
  const float* f0 = feat + (size_t)b * 3 * N_PTS;
  const float cx = newxyz[((size_t)b * 3 + 0) * S_NP + s];
  const float cy = newxyz[((size_t)b * 3 + 1) * S_NP + s];
  const float cz = newxyz[((size_t)b * 3 + 2) * S_NP + s];
  const size_t jbase = (size_t)gw * K_S;
  int cnt = 0, first_p = -1;
  for (int ch = 0; ch < N_PTS / 32 && cnt < K_S; ++ch) {
    const int p = ch * 32 + lane;
    const float dx = px[p] - cx, dy = py[p] - cy, dz = pz[p] - cz;
    const bool in = (dx * dx + dy * dy + dz * dz) <= R2;
    const unsigned mask = (unsigned)__ballot(in);      // wave32: low 32 bits
    if (first_p < 0 && mask) first_p = ch * 32 + __builtin_ctz(mask);
    const int pos = cnt + __builtin_popcount(mask & ((1u << lane) - 1u));
    if (in && pos < K_S) {
      v16h vals = {};
      vals[0] = (_Float16)dx; vals[1] = (_Float16)dy; vals[2] = (_Float16)dz;
      vals[3] = (_Float16)f0[0 * (size_t)N_PTS + p];
      vals[4] = (_Float16)f0[1 * (size_t)N_PTS + p];
      vals[5] = (_Float16)f0[2 * (size_t)N_PTS + p];
      *(v16h*)(x0 + (jbase + pos) * 16) = vals;
    }
    cnt += __builtin_popcount(mask);
  }
  const int filled = cnt < K_S ? cnt : K_S;
  if (filled < K_S) {
    v16h vals = {};
    if (first_p >= 0) {
      const int p = first_p;
      vals[0] = (_Float16)(px[p] - cx); vals[1] = (_Float16)(py[p] - cy);
      vals[2] = (_Float16)(pz[p] - cz);
      vals[3] = (_Float16)f0[0 * (size_t)N_PTS + p];
      vals[4] = (_Float16)f0[1 * (size_t)N_PTS + p];
      vals[5] = (_Float16)f0[2 * (size_t)N_PTS + p];
    }
    for (int slot = filled + lane; slot < K_S; slot += 32)
      *(v16h*)(x0 + (jbase + slot) * 16) = vals;
  }
}

// ---------------------------------------------------------------------------
// 3) WMMA GEMM, fully compile-time shapes.
//    Z[j][COUT] = W[COUT][CIN] * X[j][CP]  (CP = padded channel stride).
//    f16 in/out, f32 accum via v_wmma_f32_16x16x32_f16.
//    BN: per-lane scale/shift K-slices hoisted to registers; packed f16
//    fma + relu on the 32-byte B fragment right after its 2x b128 load.
//    A/B fragments use contiguous K order k = 16*hi + e (wave32); D per ISA layout.
// ---------------------------------------------------------------------------
template<int CIN, int CP, int COUT, bool BN>
__global__ __launch_bounds__(256) void gemm_wmma_kernel(const _Float16* __restrict__ X,
                                                        const float* __restrict__ ab,
                                                        const float* __restrict__ W,
                                                        _Float16* __restrict__ Z) {
  constexpr int KSTEPS = (CP + 31) / 32;
  constexpr int MT = COUT / 16;
  const int lane = threadIdx.x & 31;
  const int gw   = blockIdx.x * 8 + (threadIdx.x >> 5);
  const int mtile = gw % MT;
  const int totalWaves = gridDim.x * 8;
  const int ntStep = totalWaves / MT;
  const int hi = (lane >> 4) & 1;
  const int ml = lane & 15;

  // A fragments (weights, gathered once). lanes 0-15: M=lane,
  // K 0..15; lanes 16-31: K 16..31 (matches B).
  v16h afrag[KSTEPS];
#pragma unroll
  for (int ks = 0; ks < KSTEPS; ++ks) {
    v16h a = {};
#pragma unroll
    for (int v = 0; v < 8; ++v) {
      const int kb = hi * 16 + 2 * v;
#pragma unroll
      for (int e = 0; e < 2; ++e) {
        const int k = ks * 32 + kb + e;
        const float w = (k < CIN) ? W[(size_t)(mtile * 16 + ml) * CIN + k] : 0.0f;
        a[2 * v + e] = (_Float16)w;
      }
    }
    afrag[ks] = a;
  }

  // Per-lane BN scale/shift for this lane's fixed K-slice (channels
  // ks*32 + hi*16 + 0..15), converted to f16 for packed math.
  v16h av[KSTEPS], sv[KSTEPS];
  if (BN) {
#pragma unroll
    for (int ks = 0; ks < KSTEPS; ++ks) {
#pragma unroll
      for (int e = 0; e < 16; ++e) {
        const int k = ks * 32 + hi * 16 + e;
        av[ks][e] = (_Float16)ab[k];
        sv[ks][e] = (_Float16)ab[128 + k];
      }
    }
  }

  for (int nt = gw / MT; nt < NT_TILES; nt += ntStep) {
    const size_t col = (size_t)nt * 16 + ml;
    if (nt + ntStep < NT_TILES)   // gfx1250 global prefetch of next B tile
      __builtin_prefetch(X + (col + (size_t)ntStep * 16) * CP, 0, 1);
    v8f acc = {};
#pragma unroll
    for (int ks = 0; ks < KSTEPS; ++ks) {
      // B layout: lanes 0-15 column=lane, K 0..15; lanes 16-31 K 16..31.
      v16h bf = {};
      if (ks * 32 + 16 <= CP || hi == 0) {        // CP=16: hi lanes load column col+1; A is 0 there
        bf = *(const v16h*)(X + col * CP + ks * 32 + hi * 16); // 2x b128
        if (BN) {
          bf = bf * av[ks] + sv[ks];              // packed f16 fma
#pragma unroll
          for (int e = 0; e < 16; ++e) {          // packed relu
            const _Float16 t = bf[e];
            bf[e] = t > (_Float16)0.f ? t : (_Float16)0.f;
          }
        }
      }
      acc = __builtin_amdgcn_wmma_f32_16x16x32_f16(false, afrag[ks], false, bf,
                                                   (short)0, acc, false, false);
    }
    // D layout: VGPR v, lane half -> M = mtile*16 + hi*8 + v; N = ml.
    // Contiguous 8 halves -> single 16B store.
    v8h dv;
#pragma unroll
    for (int v = 0; v < 8; ++v) dv[v] = (_Float16)acc[v];
    *(v8h*)(Z + col * COUT + mtile * 16 + hi * 8) = dv;
  }
}

// ---------------------------------------------------------------------------
// 4) Deterministic hierarchical per-channel sum / sum^2 on [j][C] layout.
//    Stage 1: 256 blocks write per-block partials; stage 2 inside
//    bn_finalize sums the 256 partials in fixed order.
// ---------------------------------------------------------------------------
template<int C>
__global__ __launch_bounds__(256) void channel_reduce_kernel(const _Float16* __restrict__ Z,
                                                             float* __restrict__ partial) {
  constexpr int SUB = 256 / C;
  __shared__ float ssum[256], ssq[256];
  const int tid = threadIdx.x;
  const int c = tid % C;
  const int sub = tid / C;
  const size_t jstart = (size_t)blockIdx.x * (NTOT / RED_BLOCKS);
  float s = 0.f, q = 0.f;
  for (size_t j = jstart + sub; j < jstart + NTOT / RED_BLOCKS; j += SUB) {
    const float v = (float)Z[j * C + c];          // coalesced across lanes
    s += v; q += v * v;
  }
  ssum[tid] = s; ssq[tid] = q;
  __syncthreads();
  if (tid < C) {
    float ts = 0.f, tq = 0.f;
#pragma unroll
    for (int u = 0; u < SUB; ++u) { ts += ssum[c + u * C]; tq += ssq[c + u * C]; }
    partial[(size_t)blockIdx.x * 256 + c]       = ts;
    partial[(size_t)blockIdx.x * 256 + 128 + c] = tq;
  }
}

// BN affine fold: a = g*rsqrt(var+eps), s = beta - mean*a.
// (the conv bias is a per-channel constant -> cancels inside BN)
__global__ void bn_finalize_kernel(const float* __restrict__ partial,
                                   const float* __restrict__ g,
                                   const float* __restrict__ beta,
                                   float* __restrict__ ab, int C) {
  const int c = threadIdx.x;
  if (c < C) {
    float s = 0.f, q = 0.f;
    for (int i = 0; i < RED_BLOCKS; ++i) {        // fixed order: deterministic
      s += partial[(size_t)i * 256 + c];
      q += partial[(size_t)i * 256 + 128 + c];
    }
    const float n    = (float)NTOT;
    const float mean = s / n;
    const float var  = q / n - mean * mean;
    const float a    = g[c] * rsqrtf(var + BN_EPS);
    ab[c]       = a;
    ab[128 + c] = beta[c] - mean * a;
  }
}

// ---------------------------------------------------------------------------
// 5) Final BN+ReLU+max over K=32 on [j][128] layout, LDS-transposed so both
//    the z3 reads (c contiguous) and the out writes (s contiguous) coalesce.
//    One block per (b, 32-wide s strip).
// ---------------------------------------------------------------------------
__global__ __launch_bounds__(256) void maxout_kernel(const _Float16* __restrict__ Z,
                                                     const float* __restrict__ ab,
                                                     float* __restrict__ out) {
  __shared__ float tile[128 * 33];
  const int b  = blockIdx.x >> 5;
  const int s0 = (blockIdx.x & 31) * 32;
  const int tid = threadIdx.x;
  for (int it = tid; it < 128 * 32; it += 256) {
    const int c = it & 127;                        // contiguous across lanes
    const int sidx = it >> 7;
    const float a  = ab[c];
    const float sh = ab[128 + c];
    const size_t base = ((size_t)(b * S_NP + s0 + sidx) * K_S) * 128 + c;
    float m = -1e30f;
#pragma unroll 4
    for (int k = 0; k < K_S; ++k) {
      float v = (float)Z[base + (size_t)k * 128] * a + sh;
      v = v > 0.f ? v : 0.f;
      m = fmaxf(m, v);
    }
    tile[c * 33 + sidx] = m;
  }
  __syncthreads();
  for (int it = tid; it < 128 * 32; it += 256) {
    const int sc = it & 31;                        // contiguous across lanes
    const int r  = it >> 5;
    out[(size_t)BATCH * 3 * S_NP + ((size_t)b * 128 + r) * S_NP + s0 + sc] =
        tile[r * 33 + sc];
  }
}

// ---------------------------------------------------------------------------
extern "C" void kernel_launch(void* const* d_in, const int* in_sizes, int n_in,
                              void* d_out, int out_size, void* d_ws, size_t ws_size,
                              hipStream_t stream) {
  const float* xyz  = (const float*)d_in[0];
  const float* feat = (const float*)d_in[1];
  const float* w0 = (const float*)d_in[2];
  const float* g0 = (const float*)d_in[4];
  const float* bt0 = (const float*)d_in[5];
  const float* w1 = (const float*)d_in[6];
  const float* g1 = (const float*)d_in[8];
  const float* bt1 = (const float*)d_in[9];
  const float* w2 = (const float*)d_in[10];
  const float* g2 = (const float*)d_in[12];
  const float* bt2 = (const float*)d_in[13];
  float* out = (float*)d_out;

  char* ws = (char*)d_ws;
  const size_t x0B = NTOT * 16  * sizeof(_Float16);   // [j][16]
  const size_t z1B = NTOT * 64  * sizeof(_Float16);   // [j][64]
  const size_t z2B = NTOT * 64  * sizeof(_Float16);
  const size_t z3B = NTOT * 128 * sizeof(_Float16);   // [j][128]
  _Float16* x0 = (_Float16*)ws;
  _Float16* z1 = (_Float16*)(ws + x0B);
  _Float16* z2 = (_Float16*)(ws + x0B + z1B);
  _Float16* z3 = (_Float16*)(ws + x0B + z1B + z2B);
  float* partial = (float*)(ws + x0B + z1B + z2B + z3B);  // 256*256 f32
  float* ab      = partial + RED_BLOCKS * 256;            // 256 f32

  fps_kernel<<<BATCH, 256, 0, stream>>>(xyz, out);
  ball_group_kernel<<<(BATCH * S_NP) / 8, 256, 0, stream>>>(xyz, feat, out, x0);

  // layer 1: 6 -> 64 (raw input, CP=16 zero-padded)
  gemm_wmma_kernel<6, 16, 64, false><<<512, 256, 0, stream>>>(x0, nullptr, w0, z1);
  channel_reduce_kernel<64><<<RED_BLOCKS, 256, 0, stream>>>(z1, partial);
  bn_finalize_kernel<<<1, 128, 0, stream>>>(partial, g0, bt0, ab, 64);

  // layer 2: 64 -> 64 (BN+ReLU of layer1 fused into B-fragment loads)
  gemm_wmma_kernel<64, 64, 64, true><<<512, 256, 0, stream>>>(z1, ab, w1, z2);
  channel_reduce_kernel<64><<<RED_BLOCKS, 256, 0, stream>>>(z2, partial);
  bn_finalize_kernel<<<1, 128, 0, stream>>>(partial, g1, bt1, ab, 64);

  // layer 3: 64 -> 128
  gemm_wmma_kernel<64, 64, 128, true><<<512, 256, 0, stream>>>(z2, ab, w2, z3);
  channel_reduce_kernel<128><<<RED_BLOCKS, 256, 0, stream>>>(z3, partial);
  bn_finalize_kernel<<<1, 128, 0, stream>>>(partial, g2, bt2, ab, 128);

  // BN+ReLU + max over nsample -> new_features [B,128,S]
  maxout_kernel<<<BATCH * (S_NP / 32), 256, 0, stream>>>(z3, ab, out);
}
